// GraphSAGEModel_24103356465366
// MI455X (gfx1250) — matrix operand order repaired and checked by reading, not compiled, not run on hardware
//
#include <hip/hip_runtime.h>

typedef __attribute__((ext_vector_type(16))) __bf16 v16bf;
typedef __attribute__((ext_vector_type(8)))  float  v8f;

#define D_FEAT 256

// ---------------- utility kernels ----------------

__global__ __launch_bounds__(256) void fill0_kernel(float4* __restrict__ p, int n4) {
  int tid = blockIdx.x * 256 + threadIdx.x;
  if (tid < n4) p[tid] = make_float4(0.f, 0.f, 0.f, 0.f);
}

__global__ __launch_bounds__(256) void deg_kernel(const int* __restrict__ dst,
                                                  float* __restrict__ deg, int nE) {
  int tid = blockIdx.x * 256 + threadIdx.x;
  if (tid < nE) atomicAdd(&deg[dst[tid]], 1.0f);
}

__global__ __launch_bounds__(256) void invdeg_kernel(float* __restrict__ deg, int n) {
  int tid = blockIdx.x * 256 + threadIdx.x;
  if (tid < n) deg[tid] = 1.0f / fmaxf(deg[tid], 1.0f);
}

// segment-sum: 64 threads per edge, float4 per thread (256 features)
__global__ __launch_bounds__(256) void scatter_kernel(const float* __restrict__ H,
                                                      const int* __restrict__ src,
                                                      const int* __restrict__ dst,
                                                      float* __restrict__ agg, int nE) {
  int tid = blockIdx.x * 256 + threadIdx.x;
  int e = tid >> 6;
  if (e >= nE) return;
  int c = (tid & 63) << 2;
  int s = src[e], d = dst[e];
  float4 v = *(const float4*)(H + (size_t)s * D_FEAT + c);
  float* p = agg + (size_t)d * D_FEAT + c;
  atomicAdd(p + 0, v.x);
  atomicAdd(p + 1, v.y);
  atomicAdd(p + 2, v.z);
  atomicAdd(p + 3, v.w);
}

// fp32 -> bf16 hi + bf16 residual lo planes; optional per-row scale (mean = sum * invdeg)
__global__ __launch_bounds__(256) void cvt_kernel(const float* __restrict__ in,
                                                  const float* __restrict__ rowscale,
                                                  __bf16* __restrict__ hi,
                                                  __bf16* __restrict__ lo, int n) {
  int tid = blockIdx.x * 256 + threadIdx.x;
  if (tid >= n) return;
  float v = in[tid];
  if (rowscale) v *= rowscale[tid >> 8];   // 256 features per row
  __bf16 h = (__bf16)v;
  hi[tid] = h;
  lo[tid] = (__bf16)(v - (float)h);
}

// pack fp32 weight [K=256 x dout] into WMMA B-fragment order, hi/lo bf16 planes.
// fragment(kt, nt): lane L holds B[kt*32 + (L>>4)*8 + j + (j&8)][nt*16 + (L&15)], j=0..15
__global__ __launch_bounds__(256) void packw_kernel(const float* __restrict__ W,
                                                    __bf16* __restrict__ Phi,
                                                    __bf16* __restrict__ Plo,
                                                    int dout, int ntiles) {
  int tid = blockIdx.x * 256 + threadIdx.x;
  int total = 8 * ntiles * 32;
  if (tid >= total) return;
  int lane = tid & 31;
  int nt = (tid >> 5) % ntiles;
  int kt = (tid >> 5) / ntiles;
  int n = nt * 16 + (lane & 15);
  int kbase = kt * 32 + (lane >> 4) * 8;
  size_t base = (size_t)tid * 16;
  for (int j = 0; j < 16; ++j) {
    float v = (n < dout) ? W[(size_t)(kbase + j + (j & 8)) * dout + n] : 0.f;
    __bf16 h = (__bf16)v;
    Phi[base + j] = h;
    Plo[base + j] = (__bf16)(v - (float)h);
  }
}

union FragA { uint4 u[2]; v16bf v; };

// out = relu?( mean @ Wl + H @ Wr + b ), split-bf16 (hi*hi + hi*lo + lo*hi) in f32 acc.
// One wave per 16x16 C tile; 4 waves/block.
__global__ __launch_bounds__(128) void sage_gemm_kernel(
    const __bf16* __restrict__ Ahi, const __bf16* __restrict__ Alo,
    const __bf16* __restrict__ Hhi, const __bf16* __restrict__ Hlo,
    const __bf16* __restrict__ BlHi, const __bf16* __restrict__ BlLo,
    const __bf16* __restrict__ BrHi, const __bf16* __restrict__ BrLo,
    const float* __restrict__ bias, float* __restrict__ out,
    int nrows, int rowTiles, int ntiles, int dout, int doRelu) {
  int wave = blockIdx.x * 4 + (threadIdx.x >> 5);
  int lane = threadIdx.x & 31;
  if (wave >= rowTiles * ntiles) return;       // wave-uniform guard (EXEC stays all-1s)
  int tn = wave % ntiles;
  int tm = wave / ntiles;

  int mrow = tm * 16 + (lane & 15);
  if (mrow >= nrows) mrow = nrows - 1;          // clamp loads; stores guarded below
  int kgrp8 = (lane >> 4) * 8;
  size_t aoff = (size_t)mrow * D_FEAT;

  v8f acc = {0.f, 0.f, 0.f, 0.f, 0.f, 0.f, 0.f, 0.f};

#pragma unroll
  for (int kt = 0; kt < 8; ++kt) {
    int k0 = kt * 32 + kgrp8;
    size_t boff = ((size_t)(kt * ntiles + tn) * 32 + lane) * 16;

    // ---- pair 1: mean @ Wl ----
    FragA ah, al;
    ah.u[0] = *(const uint4*)(Ahi + aoff + k0);
    ah.u[1] = *(const uint4*)(Ahi + aoff + k0 + 16);
    al.u[0] = *(const uint4*)(Alo + aoff + k0);
    al.u[1] = *(const uint4*)(Alo + aoff + k0 + 16);
    v16bf bh = *(const v16bf*)(BlHi + boff);
    v16bf bl = *(const v16bf*)(BlLo + boff);
    acc = __builtin_amdgcn_wmma_f32_16x16x32_bf16(false, ah.v, false, bh, (short)0, acc, false, false);
    acc = __builtin_amdgcn_wmma_f32_16x16x32_bf16(false, al.v, false, bh, (short)0, acc, false, false);
    acc = __builtin_amdgcn_wmma_f32_16x16x32_bf16(false, ah.v, false, bl, (short)0, acc, false, false);

    // ---- pair 2: H @ Wr ----
    ah.u[0] = *(const uint4*)(Hhi + aoff + k0);
    ah.u[1] = *(const uint4*)(Hhi + aoff + k0 + 16);
    al.u[0] = *(const uint4*)(Hlo + aoff + k0);
    al.u[1] = *(const uint4*)(Hlo + aoff + k0 + 16);
    bh = *(const v16bf*)(BrHi + boff);
    bl = *(const v16bf*)(BrLo + boff);
    acc = __builtin_amdgcn_wmma_f32_16x16x32_bf16(false, ah.v, false, bh, (short)0, acc, false, false);
    acc = __builtin_amdgcn_wmma_f32_16x16x32_bf16(false, al.v, false, bh, (short)0, acc, false, false);
    acc = __builtin_amdgcn_wmma_f32_16x16x32_bf16(false, ah.v, false, bl, (short)0, acc, false, false);
  }

  // epilogue: C/D layout -> lane L, vgpr r holds C[r + 8*(L>>4)][L&15]
  int col = tn * 16 + (lane & 15);
  int rbase = tm * 16 + ((lane >> 4) << 3);
  if (col < dout) {
    float bv = bias[col];
#pragma unroll
    for (int r = 0; r < 8; ++r) {
      int row = rbase + r;
      if (row < nrows) {
        float v = acc[r] + bv;
        if (doRelu) v = fmaxf(v, 0.f);
        out[(size_t)row * dout + col] = v;
      }
    }
  }
}

// one wave per row of 40 logits
__global__ __launch_bounds__(256) void logsoftmax_kernel(const float* __restrict__ logits,
                                                         float* __restrict__ out, int nrows) {
  int wid = blockIdx.x * 8 + (threadIdx.x >> 5);
  int lane = threadIdx.x & 31;
  if (wid >= nrows) return;
  const float* row = logits + (size_t)wid * 40;
  float v0 = row[lane];
  float v1 = (lane < 8) ? row[32 + lane] : -3.4e38f;
  float m = fmaxf(v0, v1);
  for (int off = 16; off > 0; off >>= 1) m = fmaxf(m, __shfl_xor(m, off, 32));
  float s = expf(v0 - m) + ((lane < 8) ? expf(v1 - m) : 0.f);
  for (int off = 16; off > 0; off >>= 1) s += __shfl_xor(s, off, 32);
  float lse = m + logf(s);
  out[(size_t)wid * 40 + lane] = v0 - lse;
  if (lane < 8) out[(size_t)wid * 40 + 32 + lane] = v1 - lse;
}

// ---------------- host launcher ----------------

extern "C" void kernel_launch(void* const* d_in, const int* in_sizes, int n_in,
                              void* d_out, int out_size, void* d_ws, size_t ws_size,
                              hipStream_t stream) {
  (void)n_in; (void)out_size; (void)ws_size;
  const float* x = (const float*)d_in[0];
  const int* ei = (const int*)d_in[1];
  const float* Wl[3] = {(const float*)d_in[2], (const float*)d_in[5], (const float*)d_in[8]};
  const float* Wr[3] = {(const float*)d_in[3], (const float*)d_in[6], (const float*)d_in[9]};
  const float* bb[3] = {(const float*)d_in[4], (const float*)d_in[7], (const float*)d_in[10]};
  const int N = in_sizes[0] / D_FEAT;
  const int E = in_sizes[1] / 2;
  const int* src = ei;
  const int* dst = ei + E;

  char* ws = (char*)d_ws;
  size_t off = 0;
  auto take = [&](size_t bytes) -> char* {
    char* p = ws + off;
    off += (bytes + 255) & ~(size_t)255;
    return p;
  };
  const size_t nFeat = (size_t)N * D_FEAT;
  float*  deg    = (float*)take((size_t)N * 4);
  float*  bufA   = (float*)take(nFeat * 4);
  float*  bufB   = (float*)take(nFeat * 4);
  __bf16* Ahi    = (__bf16*)take(nFeat * 2);
  __bf16* Alo    = (__bf16*)take(nFeat * 2);
  __bf16* Hhi    = (__bf16*)take(nFeat * 2);
  __bf16* Hlo    = (__bf16*)take(nFeat * 2);
  float*  logits = (float*)take((size_t)N * 40 * 4);

  const int ntl[3] = {16, 16, 3};
  const int dol[3] = {256, 256, 40};
  __bf16 *BlHi[3], *BlLo[3], *BrHi[3], *BrLo[3];
  for (int l = 0; l < 3; ++l) {
    size_t pb = (size_t)8 * ntl[l] * 32 * 16 * 2;   // bytes per plane
    BlHi[l] = (__bf16*)take(pb); BlLo[l] = (__bf16*)take(pb);
    BrHi[l] = (__bf16*)take(pb); BrLo[l] = (__bf16*)take(pb);
  }

  const int nF = (int)nFeat;
  const int rowTiles = (N + 15) / 16;

  // degrees -> 1/max(deg,1)
  fill0_kernel<<<(N / 4 + 255) / 256, 256, 0, stream>>>((float4*)deg, N / 4);
  deg_kernel<<<(E + 255) / 256, 256, 0, stream>>>(dst, deg, E);
  invdeg_kernel<<<(N + 255) / 256, 256, 0, stream>>>(deg, N);

  // pack weights into fragment order (tiny)
  for (int l = 0; l < 3; ++l) {
    int threads = 8 * ntl[l] * 32;
    packw_kernel<<<(threads + 255) / 256, 256, 0, stream>>>(Wl[l], BlHi[l], BlLo[l], dol[l], ntl[l]);
    packw_kernel<<<(threads + 255) / 256, 256, 0, stream>>>(Wr[l], BrHi[l], BrLo[l], dol[l], ntl[l]);
  }

  auto run_layer = [&](const float* Hin, float* aggBuf, float* outBuf, int l, int doRelu) {
    fill0_kernel<<<(nF / 4 + 255) / 256, 256, 0, stream>>>((float4*)aggBuf, nF / 4);
    scatter_kernel<<<(E * 64 + 255) / 256, 256, 0, stream>>>(Hin, src, dst, aggBuf, E);
    cvt_kernel<<<(nF + 255) / 256, 256, 0, stream>>>(aggBuf, deg, Ahi, Alo, nF);
    cvt_kernel<<<(nF + 255) / 256, 256, 0, stream>>>(Hin, nullptr, Hhi, Hlo, nF);
    int tiles = rowTiles * ntl[l];
    sage_gemm_kernel<<<(tiles + 3) / 4, 128, 0, stream>>>(
        Ahi, Alo, Hhi, Hlo, BlHi[l], BlLo[l], BrHi[l], BrLo[l],
        bb[l], outBuf, N, rowTiles, ntl[l], dol[l], doRelu);
  };

  run_layer(x,    bufA, bufB,   0, 1);
  run_layer(bufB, bufA, bufA,   1, 1);   // agg fp32 consumed by cvt before GEMM overwrites bufA
  run_layer(bufA, bufB, logits, 2, 0);

  logsoftmax_kernel<<<(N + 7) / 8, 256, 0, stream>>>(logits, (float*)d_out, N);
}
